// Attention_67740224192767
// MI455X (gfx1250) — matrix operand order repaired and checked by reading, not compiled, not run on hardware
//
#include <hip/hip_runtime.h>

// ---------------------------------------------------------------------------
// Attention layer for MI455X (gfx1250, wave32, WMMA, async-to-LDS).
//   B=8, S=1024, D=1024, H=16, hd=64, block-causal mask (8-wide), RoPE.
// Pipeline:
//   1) fp32 -> f16 converts (x, wq, wk, wv, wo)
//   2) projection GEMMs: 32Mx64N per wave (8 WMMA/K-step), weights staged in
//      LDS via global_load_async_to_lds_b128 (double-buffered, ASYNCcnt),
//      A-fragments register-prefetched one K-step ahead.
//      RoPE fused for Q/K; V stored transposed [b,h,d,s] (packed b128 stores).
//   3) flash-attention: one wave per 16-row q-tile, 32-key chunks, online
//      softmax with wave32 half-shuffles. V-fragment loads issue after the
//      score WMMAs (K-frags dead) and hide behind softmax VALU work.
//   4) output projection GEMM (same core), fp32 store to d_out.
// ---------------------------------------------------------------------------

typedef __attribute__((ext_vector_type(16))) _Float16 v16h;
typedef __attribute__((ext_vector_type(8)))  _Float16 v8h;
typedef __attribute__((ext_vector_type(8)))  float    v8f;
typedef __attribute__((ext_vector_type(4)))  _Float16 v4h;

namespace {
constexpr int kB  = 8;
constexpr int kS  = 1024;
constexpr int kD  = 1024;
constexpr int kH  = 16;
constexpr int kHD = 64;
constexpr int kM  = kB * kS;            // 8192 rows
constexpr float kScale = 0.125f;        // 1/sqrt(64)
}

// Load a 16-element f16 A/B fragment as two aligned b128 loads.
static __device__ __forceinline__ v16h ldfrag(const _Float16* p0, const _Float16* p1) {
    union { v16h v; uint4 u[2]; } f;
    f.u[0] = *(const uint4*)p0;
    f.u[1] = *(const uint4*)p1;
    return f.v;
}

static __device__ __forceinline__ v8f wmma_f16(v16h a, v16h b, v8f c) {
    return __builtin_amdgcn_wmma_f32_16x16x32_f16(false, a, false, b, (short)0, c, false, false);
}

// gfx1250 async global->LDS copy (16B per lane), tracked by ASYNCcnt.
static __device__ __forceinline__ void async_b128_to_lds(unsigned lds_off, const void* gptr) {
    unsigned long long g = (unsigned long long)(uintptr_t)gptr;
    asm volatile("global_load_async_to_lds_b128 %0, %1, off"
                 :: "v"(lds_off), "v"(g) : "memory");
}
static __device__ __forceinline__ void wait_async0() {
    asm volatile("s_wait_asynccnt 0x0" ::: "memory");
}

// ---------------------------------------------------------------------------
// fp32 -> f16 conversion, 4 elements per thread
// ---------------------------------------------------------------------------
__global__ void cvt_f32_to_f16_kernel(const float4* __restrict__ src,
                                      _Float16* __restrict__ dst, int n4) {
    int i = blockIdx.x * blockDim.x + threadIdx.x;
    if (i < n4) {
        float4 v = src[i];
        v4h o = {(_Float16)v.x, (_Float16)v.y, (_Float16)v.z, (_Float16)v.w};
        *(v4h*)(dst + 4 * (size_t)i) = o;
    }
}

// ---------------------------------------------------------------------------
// GEMM core: Y = X @ W^T   (X:[8192,1024] f16, W:[1024,1024] f16)
// Block = 4 waves = 128Mx64N; wave = 32Mx64N (8 WMMA per 32-deep K-step).
// W tile (64 cols x 32 k = 4KB) staged in LDS via async-to-LDS, double buffer.
// MODE 0: RoPE epilogue, f16 store to dst[b][h][s][d]      (Q, K)
// MODE 1: f16 packed-b128 store transposed to dst[b][h][d][s]   (V)
// MODE 2: fp32 store to dst[m][n]                          (final out)
// ---------------------------------------------------------------------------
template <int MODE>
__global__ __launch_bounds__(128, 1)
void gemm_kernel(const _Float16* __restrict__ X,
                 const _Float16* __restrict__ W,
                 const float* __restrict__ cosb,
                 const float* __restrict__ sinb,
                 void* __restrict__ dstv) {
    __shared__ _Float16 Bl[2][64][32];   // [buf][n-col][k]  (4KB per buffer)

    const int tid  = threadIdx.x;
    const int lane = tid & 31;
    const int wave = tid >> 5;
    const int h0   = lane >> 4;          // half-wave (CDNA5 frag layout)
    const int nn   = lane & 15;
    const int mblk = blockIdx.x >> 4;    // 64 macro rows of 128
    const int nblk = blockIdx.x & 15;    // 16 macro cols of 64
    const int nbase = nblk * 64;
    const int mbase = mblk * 128 + wave * 32;

    // Cooperative B-fill assignment: thread -> (column, 16-wide k-half).
    const int fcol = tid >> 1;           // 0..63
    const int fk   = (tid & 1) * 16;     // 0 or 16
    const _Float16* wsrc = W + (size_t)(nbase + fcol) * kD + fk;

    // Prologue: fill buffer 0 for kk=0.
    {
        unsigned l0 = (unsigned)(uintptr_t)&Bl[0][fcol][fk];
        async_b128_to_lds(l0,      wsrc);
        async_b128_to_lds(l0 + 16, wsrc + 8);
    }

    const _Float16* arow0 = X + (size_t)(mbase + nn) * kD;
    const _Float16* arow1 = X + (size_t)(mbase + 16 + nn) * kD;
    v16h a0 = ldfrag(arow0 + 8 * h0, arow0 + 16 + 8 * h0);
    v16h a1 = ldfrag(arow1 + 8 * h0, arow1 + 16 + 8 * h0);

    v8f acc[2][4];
    {
        v8f z = {};
        #pragma unroll
        for (int f = 0; f < 2; ++f)
            #pragma unroll
            for (int j = 0; j < 4; ++j) acc[f][j] = z;
    }

    wait_async0();
    __syncthreads();

    for (int i = 0; i < kD / 32; ++i) {
        const int kk  = i * 32;
        const int buf = i & 1;

        // Issue next B tile fill into the other buffer (overlaps compute).
        if (i + 1 < kD / 32) {
            unsigned l0 = (unsigned)(uintptr_t)&Bl[buf ^ 1][fcol][fk];
            async_b128_to_lds(l0,      wsrc + kk + 32);
            async_b128_to_lds(l0 + 16, wsrc + kk + 40);
        }
        // Prefetch next A fragments into registers.
        v16h na0 = a0, na1 = a1;
        if (i + 1 < kD / 32) {
            na0 = ldfrag(arow0 + kk + 32 + 8 * h0, arow0 + kk + 48 + 8 * h0);
            na1 = ldfrag(arow1 + kk + 32 + 8 * h0, arow1 + kk + 48 + 8 * h0);
        }

        // Compute on current buffer: 4 n-tiles x 2 m-subtiles = 8 WMMAs.
        #pragma unroll
        for (int j = 0; j < 4; ++j) {
            const _Float16* bp = &Bl[buf][j * 16 + nn][8 * h0];
            v16h b = ldfrag(bp, bp + 16);
            acc[0][j] = wmma_f16(a0, b, acc[0][j]);
            acc[1][j] = wmma_f16(a1, b, acc[1][j]);
        }

        wait_async0();
        __syncthreads();
        a0 = na0;
        a1 = na1;
    }

    // Epilogue
    #pragma unroll
    for (int f = 0; f < 2; ++f) {
        #pragma unroll
        for (int j = 0; j < 4; ++j) {
            const int ng = nbase + j * 16 + nn;   // global N (= h*64 + d)
            const int hh = ng >> 6;
            const int d  = ng & 63;
            if constexpr (MODE == 1) {
                // 8 accumulator rows map to 8 consecutive s: one b128 store.
                const int m0 = mbase + f * 16 + 8 * h0;
                const int bb = m0 >> 10;
                const int s0 = m0 & 1023;
                union { v8h h; uint4 u; } pk;
                #pragma unroll
                for (int r = 0; r < 8; ++r) pk.h[r] = (_Float16)acc[f][j][r];
                _Float16* vt = (_Float16*)dstv;
                *(uint4*)&vt[((size_t)((bb * kH + hh) * kHD + d)) * kS + s0] = pk.u;
            } else {
                #pragma unroll
                for (int r = 0; r < 8; ++r) {
                    const int m  = mbase + f * 16 + r + 8 * h0;  // C-frag: M = reg + 8*half
                    const int bb = m >> 10;
                    const int s  = m & 1023;
                    float v = acc[f][j][r];
                    if constexpr (MODE == 0) {
                        float v2 = __shfl_xor(v, 1);             // RoPE pair partner (d^1)
                        const int j2 = d >> 1;
                        const float cs = cosb[s * 32 + j2];
                        const float sn = sinb[s * 32 + j2];
                        v = (d & 1) ? (v2 * sn + v * cs) : (v * cs - v2 * sn);
                        ((_Float16*)dstv)[((size_t)((bb * kH + hh) * kS + s)) * kHD + d] = (_Float16)v;
                    } else {
                        ((float*)dstv)[(size_t)m * kD + ng] = v;
                    }
                }
            }
        }
    }
}

// ---------------------------------------------------------------------------
// Flash attention: one wave per (b, h, 16-row q-tile); keys in 32-wide chunks.
// Block-causal mask: allowed iff (kpos>>3) <= (qpos>>3).
// K-fragments live only through the score WMMAs; V-fragment loads issue after
// them and complete behind the softmax VALU phase (no spills, still overlap).
// ---------------------------------------------------------------------------
__global__ __launch_bounds__(128, 1)
void attn_kernel(const _Float16* __restrict__ Qh,
                 const _Float16* __restrict__ Kh,
                 const _Float16* __restrict__ Vt,
                 _Float16* __restrict__ Oh) {
    __shared__ _Float16 Pl[4][16][32];   // per-wave P staging (C-frag -> A-frag)

    const int lane = threadIdx.x & 31;
    const int wave = threadIdx.x >> 5;
    const int h0   = lane >> 4;
    const int nn   = lane & 15;
    const int gw   = blockIdx.x * 4 + wave;
    const int qt   = gw & 63;            // q-tile (64 per head)
    const int hh   = (gw >> 6) & 15;
    const int bb   = gw >> 10;

    const size_t bh = (size_t)(bb * kH + hh);
    const _Float16* Qb = Qh + bh * kS * kHD;
    const _Float16* Kb = Kh + bh * kS * kHD;
    const _Float16* Vb = Vt + bh * kHD * kS;

    const int q0 = qt * 16;
    const _Float16* qrow = Qb + (size_t)(q0 + nn) * kHD;
    const v16h aq0 = ldfrag(qrow + 16 * h0,      qrow + 16 * h0 + 8);       // hd 0..31
    const v16h aq1 = ldfrag(qrow + 32 + 16 * h0, qrow + 32 + 16 * h0 + 8);  // hd 32..63

    v8f acc[4];
    { v8f z = {}; acc[0] = z; acc[1] = z; acc[2] = z; acc[3] = z; }
    float mrun[8], lrun[8];
    #pragma unroll
    for (int r = 0; r < 8; ++r) { mrun[r] = -1e30f; lrun[r] = 0.0f; }

    const int nkc = (qt >> 1) + 1;       // 32-key chunks under the causal mask
    for (int kc = 0; kc < nkc; ++kc) {
        const int k0 = kc * 32;
        const _Float16* kr0 = Kb + (size_t)(k0 + nn) * kHD;
        const _Float16* kr1 = Kb + (size_t)(k0 + 16 + nn) * kHD;

        // K fragments + score WMMAs (K-frags die right after).
        v8f s0 = {}, s1 = {};
        {
            v16h b00 = ldfrag(kr0 + 16 * h0,      kr0 + 16 * h0 + 8);       // hd 0..31
            v16h b01 = ldfrag(kr0 + 32 + 16 * h0, kr0 + 32 + 16 * h0 + 8);  // hd 32..63
            v16h b10 = ldfrag(kr1 + 16 * h0,      kr1 + 16 * h0 + 8);
            v16h b11 = ldfrag(kr1 + 32 + 16 * h0, kr1 + 32 + 16 * h0 + 8);
            s0 = wmma_f16(aq0, b00, s0);
            s0 = wmma_f16(aq1, b01, s0);
            s1 = wmma_f16(aq0, b10, s1);
            s1 = wmma_f16(aq1, b11, s1);
        }

        // Issue V-fragment loads now; they complete behind the softmax math.
        v16h bv[4];
        #pragma unroll
        for (int dt = 0; dt < 4; ++dt) {
            const _Float16* vrow = Vb + (size_t)(dt * 16 + nn) * kS + k0 + 16 * h0;
            bv[dt] = ldfrag(vrow, vrow + 8);
        }

        // Online softmax; row reductions stay within each 16-lane half,
        // matching the C-fragment layout (M = reg + 8*h0, N = lane&15).
        float al[8];
        #pragma unroll
        for (int r = 0; r < 8; ++r) {
            const int qpos = q0 + r + 8 * h0;
            const int qb3  = qpos >> 3;
            float v0 = s0[r] * kScale;
            float v1 = s1[r] * kScale;
            if (((k0 + nn) >> 3)      > qb3) v0 = -1e30f;
            if (((k0 + 16 + nn) >> 3) > qb3) v1 = -1e30f;
            float mx = fmaxf(v0, v1);
            mx = fmaxf(mx, __shfl_xor(mx, 1));
            mx = fmaxf(mx, __shfl_xor(mx, 2));
            mx = fmaxf(mx, __shfl_xor(mx, 4));
            mx = fmaxf(mx, __shfl_xor(mx, 8));
            const float mnew = fmaxf(mrun[r], mx);
            const float a    = __expf(mrun[r] - mnew);
            const float p0   = __expf(v0 - mnew);
            const float p1   = __expf(v1 - mnew);
            Pl[wave][r + 8 * h0][nn]      = (_Float16)p0;
            Pl[wave][r + 8 * h0][16 + nn] = (_Float16)p1;
            float rs = p0 + p1;
            rs += __shfl_xor(rs, 1);
            rs += __shfl_xor(rs, 2);
            rs += __shfl_xor(rs, 4);
            rs += __shfl_xor(rs, 8);
            lrun[r] = lrun[r] * a + rs;
            mrun[r] = mnew;
            al[r]   = a;
        }

        // Rescale running accumulators, then P @ V.
        #pragma unroll
        for (int dt = 0; dt < 4; ++dt)
            #pragma unroll
            for (int r = 0; r < 8; ++r)
                acc[dt][r] *= al[r];

        // P as A-fragment from per-wave LDS tile (same-wave LDS is in-order).
        const _Float16* prow = &Pl[wave][nn][0];
        const v16h ap = ldfrag(prow + 16 * h0, prow + 16 * h0 + 8);

        #pragma unroll
        for (int dt = 0; dt < 4; ++dt)
            acc[dt] = wmma_f16(ap, bv[dt], acc[dt]);
    }

    // Epilogue: normalize and store f16 attn output [b][s][h*64+d].
    #pragma unroll
    for (int r = 0; r < 8; ++r) {
        const float inv = 1.0f / lrun[r];
        const int s = q0 + r + 8 * h0;
        const size_t base = ((size_t)bb * kS + s) * kD + hh * kHD;
        #pragma unroll
        for (int dt = 0; dt < 4; ++dt)
            Oh[base + dt * 16 + nn] = (_Float16)(acc[dt][r] * inv);
    }
}

// ---------------------------------------------------------------------------
// Host launch. Workspace layout (bytes):
//   [0,    16MB)  xh  : x as f16 [8192][1024]; later reused as attn f16 out
//   [16MB, 24MB)  whq/whk/whv/who : weights f16, 2MB each
//   [24MB, 40MB)  q_h : [b][h][s][64] f16
//   [40MB, 56MB)  k_h : [b][h][s][64] f16
//   [56MB, 72MB)  v_t : [b][h][64][s] f16
// ---------------------------------------------------------------------------
extern "C" void kernel_launch(void* const* d_in, const int* in_sizes, int n_in,
                              void* d_out, int out_size, void* d_ws, size_t ws_size,
                              hipStream_t stream) {
    const float* x  = (const float*)d_in[0];
    const float* wq = (const float*)d_in[1];
    const float* wk = (const float*)d_in[2];
    const float* wv = (const float*)d_in[3];
    const float* wo = (const float*)d_in[4];
    const float* fc = (const float*)d_in[5];
    const float* fs = (const float*)d_in[6];

    char* ws = (char*)d_ws;
    const size_t MB = 1024 * 1024;
    _Float16* xh  = (_Float16*)(ws);
    _Float16* whq = (_Float16*)(ws + 16 * MB);
    _Float16* whk = (_Float16*)(ws + 18 * MB);
    _Float16* whv = (_Float16*)(ws + 20 * MB);
    _Float16* who = (_Float16*)(ws + 22 * MB);
    _Float16* q_h = (_Float16*)(ws + 24 * MB);
    _Float16* k_h = (_Float16*)(ws + 40 * MB);
    _Float16* v_t = (_Float16*)(ws + 56 * MB);
    _Float16* ah  = xh;   // attention f16 output reuses xh after projections

    // 1) fp32 -> f16
    cvt_f32_to_f16_kernel<<<dim3(8192), dim3(256), 0, stream>>>((const float4*)x,  xh,  kM * kD / 4);
    cvt_f32_to_f16_kernel<<<dim3(1024), dim3(256), 0, stream>>>((const float4*)wq, whq, kD * kD / 4);
    cvt_f32_to_f16_kernel<<<dim3(1024), dim3(256), 0, stream>>>((const float4*)wk, whk, kD * kD / 4);
    cvt_f32_to_f16_kernel<<<dim3(1024), dim3(256), 0, stream>>>((const float4*)wv, whv, kD * kD / 4);
    cvt_f32_to_f16_kernel<<<dim3(1024), dim3(256), 0, stream>>>((const float4*)wo, who, kD * kD / 4);

    // 2) projections (WMMA + async-LDS weight staging); RoPE for Q/K; V transposed
    const int gemm_blocks = (kM / 128) * (kD / 64);   // 1024
    gemm_kernel<0><<<dim3(gemm_blocks), dim3(128), 0, stream>>>(xh, whq, fc, fs, (void*)q_h);
    gemm_kernel<0><<<dim3(gemm_blocks), dim3(128), 0, stream>>>(xh, whk, fc, fs, (void*)k_h);
    gemm_kernel<1><<<dim3(gemm_blocks), dim3(128), 0, stream>>>(xh, whv, fc, fs, (void*)v_t);

    // 3) flash attention: 8192 waves (one per b,h,q-tile), 4 waves/block
    attn_kernel<<<dim3(kB * kH * (kS / 16) / 4), dim3(128), 0, stream>>>(q_h, k_h, v_t, ah);

    // 4) output projection, fp32 store
    gemm_kernel<2><<<dim3(gemm_blocks), dim3(128), 0, stream>>>(ah, who, fc, fs, d_out);
}
